// ShiftWindowMSA_77060303225103
// MI455X (gfx1250) — matrix operand order repaired and checked by reading, not compiled, not run on hardware
//
#include <hip/hip_runtime.h>

// ---------------------------------------------------------------------------
// Swin shifted-window MSA, MI455X (gfx1250), wave32 + v_wmma_f32_16x16x32_bf16.
// B=16, H=W=56, C=256, WS=7 (no pad), SS=3, NH=8, HD=32, N=49 tokens/window,
// 1024 windows. All GEMMs in bf16 with fp32 accumulation (WMMA); softmax fp32.
// ---------------------------------------------------------------------------

#define WSZ     7
#define SSH     3
#define NH      8
#define HD      32
#define CD      256
#define HIMG    56
#define NWIN    64          // 8x8 windows per image
#define BATCH   16
#define BWIN    (BATCH*NWIN) // 1024 windows
#define NTOK    49
#define MPAD    64          // tokens padded to 4 WMMA M-tiles

typedef __attribute__((ext_vector_type(16))) __bf16 v16bf;
typedef __attribute__((ext_vector_type(8)))  __bf16 v8bf;
typedef __attribute__((ext_vector_type(8)))  float  v8f;

__device__ inline __bf16 f2bf(float f) {
    unsigned u = __float_as_uint(f);
    unsigned r = (u + 0x7FFFu + ((u >> 16) & 1u)) >> 16;
    unsigned short h = (unsigned short)r;
    __bf16 b;
    __builtin_memcpy(&b, &h, 2);
    return b;
}

__device__ inline v8f zero8() {
    v8f z;
#pragma unroll
    for (int i = 0; i < 8; ++i) z[i] = 0.0f;
    return z;
}

// A fragment: 16x32 bf16 tile, row-major, `tile` points at (row0,k0).
// Lane L holds row L%16; K = (L>=16 ? 8 : 0) + {0..7} and +16..23.
__device__ inline v16bf load_a(const __bf16* tile, int stride) {
    int lane = threadIdx.x & 31;
    int r    = lane & 15;
    int hi   = lane >> 4;
    const __bf16* p = tile + r * stride + hi * 8;
    v8bf lo = *(const v8bf*)(p);
    v8bf hh = *(const v8bf*)(p + 16);
    v16bf o;
#pragma unroll
    for (int i = 0; i < 8; ++i) { o[i] = lo[i]; o[i + 8] = hh[i]; }
    return o;
}

// B fragment from B^T stored row-major (N rows x K cols); `tileT` at (n0,k0).
// Lane L holds col n0+L%16; K = (L>=16 ? 8 : 0) + {0..7} and +16..23.
__device__ inline v16bf load_bT(const __bf16* tileT, int stride) {
    int lane = threadIdx.x & 31;
    int n    = lane & 15;
    int hi   = lane >> 4;
    const __bf16* p = tileT + n * stride + hi * 8;
    v8bf a0 = *(const v8bf*)(p);
    v8bf a1 = *(const v8bf*)(p + 16);
    v16bf o;
#pragma unroll
    for (int i = 0; i < 8; ++i) { o[i] = a0[i]; o[i + 8] = a1[i]; }
    return o;
}

__device__ inline v8f wmma_bf16(v16bf a, v16bf b, v8f c) {
    return __builtin_amdgcn_wmma_f32_16x16x32_bf16(
        /*neg_a=*/false, a, /*neg_b=*/false, b,
        /*c_mod=*/(short)0, c, /*reuse_a=*/false, /*reuse_b=*/false);
}

// ---------------------------------------------------------------------------
// Kernel 1: weights fp32 -> bf16, transposed to (N x K) row-major.
// ---------------------------------------------------------------------------
__global__ void cvt_weights(const float* __restrict__ wqkv,
                            const float* __restrict__ wproj,
                            __bf16* __restrict__ wqkvT,
                            __bf16* __restrict__ wprojT) {
    int idx = blockIdx.x * blockDim.x + threadIdx.x;
    const int n1 = 768 * 256;
    if (idx < n1) {
        int n = idx >> 8, k = idx & 255;
        wqkvT[idx] = f2bf(wqkv[k * 768 + n]);
    } else if (idx < n1 + 256 * 256) {
        int j = idx - n1;
        int n = j >> 8, k = j & 255;
        wprojT[j] = f2bf(wproj[k * 256 + n]);
    }
}

// ---------------------------------------------------------------------------
// Kernel 2: per-window QKV GEMM (64x256 @ 256x768) fused with shift+partition.
// 512 threads = 16 waves; wave handles 2 M-tiles x 6 N-tiles.
// q scaled by 1/sqrt(HD); v stored transposed (HD x MPAD) per head.
// ---------------------------------------------------------------------------
__global__ __launch_bounds__(512) void qkv_kernel(
    const float* __restrict__ query, const __bf16* __restrict__ wqkvT,
    const float* __restrict__ bqkv,
    __bf16* __restrict__ qbuf, __bf16* __restrict__ kbuf,
    __bf16* __restrict__ vtbuf) {
    __shared__ __attribute__((aligned(16))) __bf16 xs[MPAD * CD]; // 32 KB

    int win = blockIdx.x;
    int b   = win >> 6;
    int wid = win & 63;
    int wy  = wid >> 3, wx = wid & 7;

    for (int idx = threadIdx.x; idx < MPAD * CD; idx += blockDim.x) {
        int row = idx >> 8, col = idx & 255;
        float v = 0.0f;
        if (row < NTOK) {
            int iy = row / 7, ix = row % 7;
            int y = (wy * 7 + iy + SSH) % HIMG;   // roll(-SSH): rolled[g]=x[g+s]
            int x = (wx * 7 + ix + SSH) % HIMG;
            v = query[((size_t)b * (HIMG * HIMG) + y * HIMG + x) * CD + col];
        }
        xs[idx] = f2bf(v);
    }
    __syncthreads();

    int wave = threadIdx.x >> 5;
    int mh   = wave & 1;   // which pair of M-tiles
    int ng   = wave >> 1;  // N-tile group: tiles ng*6 .. ng*6+5

    v8f acc[2][6];
#pragma unroll
    for (int a = 0; a < 2; ++a)
#pragma unroll
        for (int t = 0; t < 6; ++t) acc[a][t] = zero8();

    for (int ks = 0; ks < 8; ++ks) {
        int k0 = ks * 32;
        v16bf afr[2];
#pragma unroll
        for (int a = 0; a < 2; ++a)
            afr[a] = load_a(xs + (mh * 2 + a) * 16 * CD + k0, CD);
#pragma unroll
        for (int t = 0; t < 6; ++t) {
            int n0 = (ng * 6 + t) * 16;
            v16bf bfr = load_bT(wqkvT + n0 * CD + k0, CD);
#pragma unroll
            for (int a = 0; a < 2; ++a)
                acc[a][t] = wmma_bf16(afr[a], bfr, acc[a][t]);
        }
    }

    int lane  = threadIdx.x & 31;
    int colo  = lane & 15;
    int rbase = (lane >> 4) * 8;
    const float scale = 0.17677669529663687f; // 1/sqrt(32)
#pragma unroll
    for (int a = 0; a < 2; ++a) {
        int m0 = (mh * 2 + a) * 16;
#pragma unroll
        for (int t = 0; t < 6; ++t) {
            int n = (ng * 6 + t) * 16 + colo;
            float bias = bqkv[n];
#pragma unroll
            for (int r = 0; r < 8; ++r) {
                int row = m0 + rbase + r;
                float v = acc[a][t][r] + bias;
                if (n < 256) {
                    int h = n >> 5, d = n & 31;
                    qbuf[(((size_t)win * NH + h) * MPAD + row) * HD + d] = f2bf(v * scale);
                } else if (n < 512) {
                    int nn = n - 256, h = nn >> 5, d = nn & 31;
                    kbuf[(((size_t)win * NH + h) * MPAD + row) * HD + d] = f2bf(v);
                } else {
                    int nn = n - 512, h = nn >> 5, d = nn & 31;
                    vtbuf[(((size_t)win * NH + h) * HD + d) * MPAD + row] = f2bf(v);
                }
            }
        }
    }
}

// ---------------------------------------------------------------------------
// Kernel 3: attention, one wave (32 threads) per (window, head).
// S = qk^T (WMMA, K=32) + rel-bias + shift-mask; softmax fp32; O = P·V (WMMA).
// ---------------------------------------------------------------------------
__global__ __launch_bounds__(32) void attn_kernel(
    const __bf16* __restrict__ qbuf, const __bf16* __restrict__ kbuf,
    const __bf16* __restrict__ vtbuf, const float* __restrict__ relTab,
    __bf16* __restrict__ obuf) {
    __shared__ float Sm[MPAD][65];                                   // padded stride
    __shared__ __attribute__((aligned(16))) __bf16 Pm[MPAD][72];
    __shared__ float bt[169 * NH];

    int blk = blockIdx.x;
    int win = blk >> 3;
    int h   = blk & 7;
    int wid = win & 63;
    int wy  = wid >> 3, wx = wid & 7;
    int lane = threadIdx.x;

    for (int i = lane; i < 169 * NH; i += 32) bt[i] = relTab[i];
    __syncthreads();

    const __bf16* qh = qbuf  + ((size_t)win * NH + h) * MPAD * HD;
    const __bf16* kh = kbuf  + ((size_t)win * NH + h) * MPAD * HD;
    const __bf16* vh = vtbuf + ((size_t)win * NH + h) * HD * MPAD;

    int colo  = lane & 15;
    int rbase = (lane >> 4) * 8;

    // ---- S = q @ k^T  (+ bias + mask) ----
#pragma unroll
    for (int mt = 0; mt < 4; ++mt) {
        v16bf a = load_a(qh + mt * 16 * HD, HD);
#pragma unroll
        for (int nt = 0; nt < 4; ++nt) {
            v16bf bb = load_bT(kh + nt * 16 * HD, HD); // B^T rows = k rows
            v8f c = zero8();
            c = wmma_bf16(a, bb, c);
            int j = nt * 16 + colo;
#pragma unroll
            for (int r = 0; r < 8; ++r) {
                int i = mt * 16 + rbase + r;
                float s = c[r];
                if (j >= NTOK) {
                    s = -1e30f;                       // padding columns
                } else if (i < NTOK) {
                    int iy = i / 7, ix = i % 7;
                    int jy = j / 7, jx = j % 7;
                    int rel = (iy - jy + 6) * 13 + (ix - jx + 6);
                    s += bt[rel * NH + h];
                    int gyi = wy * 7 + iy, gxi = wx * 7 + ix;
                    int gyj = wy * 7 + jy, gxj = wx * 7 + jx;
                    int zi = (gyi < 49 ? 0 : (gyi < 53 ? 1 : 2)) * 3 +
                             (gxi < 49 ? 0 : (gxi < 53 ? 1 : 2));
                    int zj = (gyj < 49 ? 0 : (gyj < 53 ? 1 : 2)) * 3 +
                             (gxj < 49 ? 0 : (gxj < 53 ? 1 : 2));
                    if (zi != zj) s -= 100.0f;
                }
                Sm[i][j] = s;
            }
        }
    }
    __syncthreads();

    // ---- softmax: 2 rows per lane, fp32, then write P as bf16 ----
#pragma unroll
    for (int rr = 0; rr < 2; ++rr) {
        int row = lane + rr * 32;
        float m = -1e30f;
        for (int c0 = 0; c0 < MPAD; ++c0) m = fmaxf(m, Sm[row][c0]);
        float sum = 0.0f;
        for (int c0 = 0; c0 < MPAD; ++c0) {
            float e = __expf(Sm[row][c0] - m);
            Sm[row][c0] = e;
            sum += e;
        }
        float rinv = 1.0f / sum;
        for (int c0 = 0; c0 < MPAD; ++c0)
            Pm[row][c0] = f2bf(Sm[row][c0] * rinv);
    }
    __syncthreads();

    // ---- O = P @ V : M=64, N=32 (2 tiles), K=64 (2 steps) ----
#pragma unroll
    for (int mt = 0; mt < 4; ++mt) {
        v16bf a0 = load_a(&Pm[mt * 16][0],  72);
        v16bf a1 = load_a(&Pm[mt * 16][32], 72);
#pragma unroll
        for (int nt = 0; nt < 2; ++nt) {
            v8f c = zero8();
            v16bf b0 = load_bT(vh + nt * 16 * MPAD + 0,  MPAD); // B^T = v^T rows
            c = wmma_bf16(a0, b0, c);
            v16bf b1 = load_bT(vh + nt * 16 * MPAD + 32, MPAD);
            c = wmma_bf16(a1, b1, c);
            int d0 = nt * 16 + colo;
#pragma unroll
            for (int r = 0; r < 8; ++r) {
                int row = mt * 16 + rbase + r;
                obuf[(size_t)win * MPAD * CD + row * CD + h * HD + d0] = f2bf(c[r]);
            }
        }
    }
}

// ---------------------------------------------------------------------------
// Kernel 4: output projection (64x256 @ 256x256) fused w/ reverse + un-shift.
// 256 threads = 8 waves; wave handles 4 M-tiles x 2 N-tiles.
// ---------------------------------------------------------------------------
__global__ __launch_bounds__(256) void proj_kernel(
    const __bf16* __restrict__ obuf, const __bf16* __restrict__ wprojT,
    const float* __restrict__ bproj, float* __restrict__ out) {
    __shared__ __attribute__((aligned(16))) __bf16 os[MPAD * CD]; // 32 KB

    int win = blockIdx.x;
    int b   = win >> 6;
    int wid = win & 63;
    int wy  = wid >> 3, wx = wid & 7;

    const unsigned* src = (const unsigned*)(obuf + (size_t)win * MPAD * CD);
    unsigned* dst = (unsigned*)os;
    for (int i = threadIdx.x; i < MPAD * CD / 2; i += blockDim.x) dst[i] = src[i];
    __syncthreads();

    int wave = threadIdx.x >> 5;
    int lane = threadIdx.x & 31;

    v8f acc[4][2];
#pragma unroll
    for (int mt = 0; mt < 4; ++mt)
#pragma unroll
        for (int t = 0; t < 2; ++t) acc[mt][t] = zero8();

    for (int ks = 0; ks < 8; ++ks) {
        int k0 = ks * 32;
        v16bf afr[4];
#pragma unroll
        for (int mt = 0; mt < 4; ++mt)
            afr[mt] = load_a(os + mt * 16 * CD + k0, CD);
#pragma unroll
        for (int t = 0; t < 2; ++t) {
            int n0 = (wave * 2 + t) * 16;
            v16bf bfr = load_bT(wprojT + n0 * CD + k0, CD);
#pragma unroll
            for (int mt = 0; mt < 4; ++mt)
                acc[mt][t] = wmma_bf16(afr[mt], bfr, acc[mt][t]);
        }
    }

    int colo  = lane & 15;
    int rbase = (lane >> 4) * 8;
#pragma unroll
    for (int mt = 0; mt < 4; ++mt) {
#pragma unroll
        for (int t = 0; t < 2; ++t) {
            int n = (wave * 2 + t) * 16 + colo;
            float bias = bproj[n];
#pragma unroll
            for (int r = 0; r < 8; ++r) {
                int row = mt * 16 + rbase + r;
                if (row < NTOK) {
                    int iy = row / 7, ix = row % 7;
                    int y = (wy * 7 + iy + SSH) % HIMG; // roll(+SSH) inverse map
                    int x = (wx * 7 + ix + SSH) % HIMG;
                    out[((size_t)b * (HIMG * HIMG) + y * HIMG + x) * CD + n] =
                        acc[mt][t][r] + bias;
                }
            }
        }
    }
}

// ---------------------------------------------------------------------------
extern "C" void kernel_launch(void* const* d_in, const int* in_sizes, int n_in,
                              void* d_out, int out_size, void* d_ws, size_t ws_size,
                              hipStream_t stream) {
    (void)in_sizes; (void)n_in; (void)out_size; (void)ws_size;
    const float* query = (const float*)d_in[0];
    const float* wqkv  = (const float*)d_in[1];
    const float* bqkv  = (const float*)d_in[2];
    const float* wproj = (const float*)d_in[3];
    const float* bproj = (const float*)d_in[4];
    const float* relT  = (const float*)d_in[5];

    char* ws = (char*)d_ws;
    const size_t SZ_WQKVT = 768u * 256u * 2u;           // 384 KB
    const size_t SZ_WPROJT = 256u * 256u * 2u;          // 128 KB
    const size_t SZ_QKV   = (size_t)BWIN * NH * MPAD * HD * 2u; // 32 MB each
    __bf16* wqkvT  = (__bf16*)(ws);
    __bf16* wprojT = (__bf16*)(ws + SZ_WQKVT);
    __bf16* qbuf   = (__bf16*)(ws + SZ_WQKVT + SZ_WPROJT);
    __bf16* kbuf   = (__bf16*)(ws + SZ_WQKVT + SZ_WPROJT + SZ_QKV);
    __bf16* vtbuf  = (__bf16*)(ws + SZ_WQKVT + SZ_WPROJT + 2 * SZ_QKV);
    __bf16* obuf   = (__bf16*)(ws + SZ_WQKVT + SZ_WPROJT + 3 * SZ_QKV);

    cvt_weights<<<1024, 256, 0, stream>>>(wqkv, wproj, wqkvT, wprojT);
    qkv_kernel<<<BWIN, 512, 0, stream>>>(query, wqkvT, bqkv, qbuf, kbuf, vtbuf);
    attn_kernel<<<BWIN * NH, 32, 0, stream>>>(qbuf, kbuf, vtbuf, relT, obuf);
    proj_kernel<<<BWIN, 256, 0, stream>>>(obuf, wprojT, bproj, (float*)d_out);
}
